// VectorQuantizer_25967372272007
// MI455X (gfx1250) — matrix operand order repaired and checked by reading, not compiled, not run on hardware
//
#include <hip/hip_runtime.h>
#include <hip/hip_bf16.h>

typedef __attribute__((ext_vector_type(16))) __bf16 v16bf;
typedef __attribute__((ext_vector_type(8)))  __bf16 v8bf;
typedef __attribute__((ext_vector_type(8)))  float  v8f;

#define VQ_N        32768      // b*h*w
#define VQ_D        64
#define VQ_K        8192
#define VQ_HW       4096
#define VQ_CSTRIDE  4096       // channel stride in [b,d,h,w]
#define VQ_BSTRIDE  262144     // batch stride in [b,d,h,w]
#define ROWS_PER_WG 256        // 8 waves x 32 rows
#define CHUNK       64         // codebook rows staged in LDS per buffer
#define KCHUNKS     (VQ_K / CHUNK)
#define PSTR        72         // padded LDS row stride (bf16 elems): kills ds bank conflicts
#define LOSS_POS    2097152
#define IDX_OFF     2097153

#define WMMA_BF16(A, B, C) \
    __builtin_amdgcn_wmma_f32_16x16x32_bf16(false, (A), false, (B), (short)0, (C), false, false)

__device__ __forceinline__ v16bf ldb16(const __bf16* p) {   // 16B-aligned 32B LDS read
    const v8bf lo = *(const v8bf*)p;
    const v8bf hi = *(const v8bf*)(p + 8);
    return __builtin_shufflevector(lo, hi, 0,1,2,3,4,5,6,7,8,9,10,11,12,13,14,15);
}

// ---- One-shot codebook preprocessing: split f32 -> hi/lo bf16 + exact f32 norms ----
__global__ void __launch_bounds__(256)
vq_prep(const float* __restrict__ emb, __bf16* __restrict__ ehi, __bf16* __restrict__ elo,
        float* __restrict__ enorm, float* __restrict__ loss)
{
    const int code = blockIdx.x * 256 + threadIdx.x;   // grid 32 -> 8192 codes
    if (code == 0) loss[0] = 0.f;
    const float* er = emb + code * VQ_D;
    float es = 0.f;
    #pragma unroll
    for (int j = 0; j < VQ_D; j += 8) {
        const float4 f0 = *(const float4*)(er + j);
        const float4 f1 = *(const float4*)(er + j + 4);
        const float fv[8] = { f0.x, f0.y, f0.z, f0.w, f1.x, f1.y, f1.z, f1.w };
        v8bf h, l;
        #pragma unroll
        for (int k = 0; k < 8; ++k) {
            const float f = fv[k];
            es += f * f;
            const __bf16 hh = (__bf16)f;
            h[k] = hh;
            l[k] = (__bf16)(f - (float)hh);
        }
        *(v8bf*)(ehi + code * VQ_D + j) = h;
        *(v8bf*)(elo + code * VQ_D + j) = l;
    }
    enorm[code] = es;
}

__global__ void vq_finalize(float* out, const float* loss) {
    out[LOSS_POS] = loss[0] * (1.25f / 2097152.f);   // (BETA+1) * mean((z_q - z)^2)
}

__global__ void __launch_bounds__(256)
vq_main(const float* __restrict__ z, const __bf16* __restrict__ ehi,
        const __bf16* __restrict__ elo, const float* __restrict__ enorm,
        const float* __restrict__ emb, float* __restrict__ out, float* __restrict__ loss)
{
    __shared__ __align__(16) __bf16 ehL[2][CHUNK * PSTR];
    __shared__ __align__(16) __bf16 elL[2][CHUNK * PSTR];
    __shared__ float enL[2][CHUNK];
    __shared__ int   idxL[ROWS_PER_WG];

    const int lane = threadIdx.x & 31;
    const int wv   = threadIdx.x >> 5;   // wave 0..7
    const int half = lane >> 4;
    const int m    = lane & 15;
    const int nbase = blockIdx.x * ROWS_PER_WG;

    // ---- Resident A: two 16-row tiles per wave (32 rows), split hi/lo bf16, contiguous k = 32*i + 16*half + j ----
    v16bf ah[2][2], al[2][2];
    float znormv[2][8];
    #pragma unroll
    for (int mt = 0; mt < 2; ++mt) {
        const int rowN = nbase + wv * 32 + mt * 16 + m;
        const float* zrow = z + (rowN >> 12) * VQ_BSTRIDE + (rowN & (VQ_HW - 1));
        float part = 0.f;
        #pragma unroll
        for (int i = 0; i < 2; ++i) {
            #pragma unroll
            for (int j = 0; j < 16; ++j) {
                const int k = i * 32 + half * 16 + j;
                const float f = zrow[k * VQ_CSTRIDE];
                part += f * f;
                const __bf16 h = (__bf16)f;
                ah[mt][i][j] = h;
                al[mt][i][j] = (__bf16)(f - (float)h);
            }
        }
        const float znorm = part + __shfl_xor(part, 16, 32);
        #pragma unroll
        for (int v = 0; v < 8; ++v)
            znormv[mt][v] = __shfl(znorm, v + 8 * half, 32);
    }

    float bestd[2][8];
    int   besti[2][8];
    #pragma unroll
    for (int mt = 0; mt < 2; ++mt)
        #pragma unroll
        for (int v = 0; v < 8; ++v) { bestd[mt][v] = 3.4028235e38f; besti[mt][v] = 0; }

    const uint4* ghi = (const uint4*)ehi;     // 8 uint4 per 64-ch row
    const uint4* glo = (const uint4*)elo;
    const int sc = threadIdx.x >> 2;          // staged code within chunk (4 threads/code)
    const int sq = threadIdx.x & 3;           // 32B quarter of the row

    // ---- Preload chunk 0 into buffer 0 ----
    {
        const int gb = sc * 8 + sq * 2;
        const uint4 h0 = ghi[gb], h1 = ghi[gb + 1];
        const uint4 l0 = glo[gb], l1 = glo[gb + 1];
        uint4* lh = (uint4*)&ehL[0][0] + sc * 9 + sq * 2;   // PSTR*2B = 144B = 9 uint4
        uint4* ll = (uint4*)&elL[0][0] + sc * 9 + sq * 2;
        lh[0] = h0; lh[1] = h1; ll[0] = l0; ll[1] = l1;
        if (threadIdx.x < CHUNK) enL[0][threadIdx.x] = enorm[threadIdx.x];
    }
    __syncthreads();

    // ---- Sweep codebook: double-buffered LDS, global prefetch overlapped with WMMA ----
    for (int ch = 0; ch < KCHUNKS; ++ch) {
        const int p = ch & 1;
        const bool more = (ch + 1 < KCHUNKS);
        uint4 nh0{}, nh1{}, nl0{}, nl1{};
        float nen = 0.f;
        if (more) {   // issue next chunk's global loads now; consumed after the barrier
            const int gb = ((ch + 1) * CHUNK + sc) * 8 + sq * 2;
            nh0 = ghi[gb]; nh1 = ghi[gb + 1];
            nl0 = glo[gb]; nl1 = glo[gb + 1];
            if (threadIdx.x < CHUNK) nen = enorm[(ch + 1) * CHUNK + threadIdx.x];
        }

        const __bf16* ehp = &ehL[p][0];
        const __bf16* elp = &elL[p][0];
        for (int t = 0; t < CHUNK / 16; ++t) {
            const int ncol = t * 16 + m;
            const __bf16* bhp = ehp + ncol * PSTR + half * 16;
            const __bf16* blp = elp + ncol * PSTR + half * 16;
            const v16bf bh0 = ldb16(bhp);
            const v16bf bl0 = ldb16(blp);
            const v16bf bh1 = ldb16(bhp + 32);
            const v16bf bl1 = ldb16(blp + 32);
            const float en   = enL[p][ncol];
            const int  codeg = ch * CHUNK + ncol;
            #pragma unroll
            for (int mt = 0; mt < 2; ++mt) {   // reuse B across two M tiles: 12 WMMAs / B load
                v8f acc = {};
                acc = WMMA_BF16(ah[mt][0], bh0, acc);
                acc = WMMA_BF16(al[mt][0], bh0, acc);
                acc = WMMA_BF16(ah[mt][0], bl0, acc);
                acc = WMMA_BF16(ah[mt][1], bh1, acc);
                acc = WMMA_BF16(al[mt][1], bh1, acc);
                acc = WMMA_BF16(ah[mt][1], bl1, acc);
                #pragma unroll
                for (int v = 0; v < 8; ++v) {
                    const float dist = znormv[mt][v] + en - 2.f * acc[v];
                    if (dist < bestd[mt][v]) { bestd[mt][v] = dist; besti[mt][v] = codeg; }
                }
            }
        }
        __syncthreads();
        if (more) {
            uint4* lh = (uint4*)&ehL[1 - p][0] + sc * 9 + sq * 2;
            uint4* ll = (uint4*)&elL[1 - p][0] + sc * 9 + sq * 2;
            lh[0] = nh0; lh[1] = nh1; ll[0] = nl0; ll[1] = nl1;
            if (threadIdx.x < CHUNK) enL[1 - p][threadIdx.x] = nen;
        }
        __syncthreads();
    }

    // ---- Min/argmin butterfly within each 16-lane half ----
    #pragma unroll
    for (int s = 1; s <= 8; s <<= 1) {
        #pragma unroll
        for (int mt = 0; mt < 2; ++mt) {
            #pragma unroll
            for (int v = 0; v < 8; ++v) {
                const float od = __shfl_xor(bestd[mt][v], s, 32);
                const int   oi = __shfl_xor(besti[mt][v], s, 32);
                if (od < bestd[mt][v] || (od == bestd[mt][v] && oi < besti[mt][v])) {
                    bestd[mt][v] = od; besti[mt][v] = oi;
                }
            }
        }
    }
    if (m == 0) {
        #pragma unroll
        for (int mt = 0; mt < 2; ++mt)
            #pragma unroll
            for (int v = 0; v < 8; ++v)
                idxL[wv * 32 + mt * 16 + 8 * half + v] = besti[mt][v];
    }
    __syncthreads();

    // ---- Epilogue: one row per thread; gather z_q, NCHW writes, loss, indices ----
    const int n2 = nbase + threadIdx.x;
    const int idx = idxL[threadIdx.x];
    float*       orow = out + (n2 >> 12) * VQ_BSTRIDE + (n2 & (VQ_HW - 1));
    const float* zr2  = z   + (n2 >> 12) * VQ_BSTRIDE + (n2 & (VQ_HW - 1));
    const float* er2  = emb + idx * VQ_D;
    float ls = 0.f;
    #pragma unroll 4
    for (int c = 0; c < VQ_D; ++c) {
        const float e  = er2[c];
        const float df = e - zr2[c * VQ_CSTRIDE];
        ls += df * df;
        orow[c * VQ_CSTRIDE] = e;   // z_q_st forward value == z_q
    }
    out[IDX_OFF + n2] = (float)idx;
    atomicAdd(loss, ls);
}

extern "C" void kernel_launch(void* const* d_in, const int* in_sizes, int n_in,
                              void* d_out, int out_size, void* d_ws, size_t ws_size,
                              hipStream_t stream) {
    const float* z   = (const float*)d_in[0];   // [8,64,64,64] f32
    const float* emb = (const float*)d_in[1];   // [8192,64]    f32
    float* out  = (float*)d_out;                // [z_q 2097152 | loss 1 | idx 32768]
    float* loss = (float*)d_ws;                 // ws[0]
    __bf16* ehi  = (__bf16*)((char*)d_ws + 256);
    __bf16* elo  = ehi + VQ_K * VQ_D;
    float*  enrm = (float*)(elo + VQ_K * VQ_D);

    vq_prep<<<VQ_K / 256, 256, 0, stream>>>(emb, ehi, elo, enrm, loss);
    vq_main<<<VQ_N / ROWS_PER_WG, 256, 0, stream>>>(z, ehi, elo, enrm, emb, out, loss);
    vq_finalize<<<1, 1, 0, stream>>>(out, loss);
}
